// Mamba4CTR_46059229283210
// MI455X (gfx1250) — matrix operand order repaired and checked by reading, not compiled, not run on hardware
//
#include <hip/hip_runtime.h>
#include <hip/hip_bf16.h>

typedef __attribute__((ext_vector_type(16))) _Float16 v16h;
typedef __attribute__((ext_vector_type(8)))  _Float16 v8h;
typedef __attribute__((ext_vector_type(8)))  float    v8f;

#define EDIM 64
#define SDIM 16
#define HID  256
#define NL   39

enum { ACT_NONE = 0, ACT_GELU = 1, ACT_RELU = 2, ACT_SOFTPLUS = 3 };

template<int ACT>
__device__ __forceinline__ float apply_act(float x) {
    if (ACT == ACT_GELU) {
        float x3 = x * x * x;
        return 0.5f * x * (1.f + tanhf(0.7978845608028654f * (x + 0.044715f * x3)));
    } else if (ACT == ACT_RELU) {
        return fmaxf(x, 0.f);
    } else if (ACT == ACT_SOFTPLUS) {
        return (x > 0.f) ? (x + log1pf(expf(-x))) : log1pf(expf(x));
    }
    return x;
}

// C[M,N] = act(A[M,K] @ W[N,K]^T + bias) (+ res).
// Requirements: K % 32 == 0, M % 64 == 0 (all call sites satisfy this).
// Block: 256 threads / 8 waves; block tile 64x64; each wave computes 16x32 via
// two v_wmma_f32_16x16x32_f16 per K-step.
//
// LDS holds tiles in fragment-ready layout:
//   Af[mtile][lane][pos] : A fragment halves, contiguous per lane (32B -> 2x ds_load_b128)
//   Bf[ntile][lane][pos] : B fragment halves, contiguous per lane
// Staging is destination-centric: thread (t, lane, hi) loads 8 consecutive
// floats from global (2x b128), converts, writes one 16B ds_store_b128.
template<int ACT>
__global__ __launch_bounds__(256)
void gemm_wmma(const float* __restrict__ A, const float* __restrict__ W,
               const float* __restrict__ bias, const float* __restrict__ res,
               float* __restrict__ C, int M, int N, int K)
{
    __shared__ __align__(32) _Float16 Af[4][32][16];
    __shared__ __align__(32) _Float16 Bf[4][32][16];

    const int tid  = threadIdx.x;
    const int m0   = blockIdx.y * 64;
    const int n0   = blockIdx.x * 64;
    const int lane = tid & 31;
    const int hi   = (tid >> 5) & 1;   // which 8-half group of a lane's fragment
    const int t    = tid >> 6;         // tile index 0..3 (m-tile for A, n-tile for B)
    const int mm   = lane & 15;

    // --- A staging source: row (m0 + t*16 + mm), cols akb + hi*8 + [0..7] + k0 ---
    const int akb = (lane < 16) ? 0 : 16;
    const float* arow = A + (size_t)(m0 + t * 16 + mm) * K + (akb + hi * 8);

    // --- B staging source: W row (n0 + t*16 + mm) clamped, cols bkb + hi*8 + [0..7] + k0 ---
    const int bkb = (lane < 16) ? 0 : 16;
    const int gn  = n0 + t * 16 + mm;
    const float wmask = (gn < N) ? 1.f : 0.f;
    const int gnc = (gn < N) ? gn : (N - 1);
    const float* wrow = W + (size_t)gnc * K + (bkb + hi * 8);

    // --- compute assignment ---
    const int wave = tid >> 5;
    const int wmt  = wave & 3;          // m-tile of this wave
    const int wnt  = (wave >> 2) * 2;   // first n-tile of this wave (0 or 2)

    v8f acc0 = {}; v8f acc1 = {};

    for (int k0 = 0; k0 < K; k0 += 32) {
        float4 a0 = *(const float4*)(arow + k0);
        float4 a1 = *(const float4*)(arow + k0 + 4);
        float4 w0 = *(const float4*)(wrow + k0);
        float4 w1 = *(const float4*)(wrow + k0 + 4);

        v8h av, wv;
        av[0] = (_Float16)a0.x; av[1] = (_Float16)a0.y;
        av[2] = (_Float16)a0.z; av[3] = (_Float16)a0.w;
        av[4] = (_Float16)a1.x; av[5] = (_Float16)a1.y;
        av[6] = (_Float16)a1.z; av[7] = (_Float16)a1.w;
        wv[0] = (_Float16)(w0.x * wmask); wv[1] = (_Float16)(w0.y * wmask);
        wv[2] = (_Float16)(w0.z * wmask); wv[3] = (_Float16)(w0.w * wmask);
        wv[4] = (_Float16)(w1.x * wmask); wv[5] = (_Float16)(w1.y * wmask);
        wv[6] = (_Float16)(w1.z * wmask); wv[7] = (_Float16)(w1.w * wmask);

        *(v8h*)&Af[t][lane][hi * 8] = av;
        *(v8h*)&Bf[t][lane][hi * 8] = wv;
        __syncthreads();

        v16h af = *(const v16h*)&Af[wmt][lane][0];
        v16h b0 = *(const v16h*)&Bf[wnt][lane][0];
        v16h b1 = *(const v16h*)&Bf[wnt + 1][lane][0];

        acc0 = __builtin_amdgcn_wmma_f32_16x16x32_f16(false, af, false, b0, (short)0, acc0, false, false);
        acc1 = __builtin_amdgcn_wmma_f32_16x16x32_f16(false, af, false, b1, (short)0, acc1, false, false);
        __syncthreads();
    }

    // Epilogue: C/D fragment layout — lane: n = lane&15, m-base = lane<16 ? 0 : 8
    const int nn = lane & 15;
    const int mb = (lane < 16) ? 0 : 8;
    #pragma unroll
    for (int r = 0; r < 8; ++r) {
        int row = m0 + wmt * 16 + mb + r;
        #pragma unroll
        for (int half = 0; half < 2; ++half) {
            int col = n0 + wnt * 16 + half * 16 + nn;
            if (col >= N) continue;
            float v = half ? acc1[r] : acc0[r];
            if (bias) v += bias[col];
            v = apply_act<ACT>(v);
            if (res) v += res[(size_t)row * N + col];
            C[(size_t)row * N + col] = v;
        }
    }
}

// x[b, f, :] = emb_table[f][ids[b, f]][:]
__global__ void gather_kernel(const float* __restrict__ dense_emb,
                              const float* __restrict__ sparse_emb,
                              const int* __restrict__ dfeat,
                              const int* __restrict__ sfeat,
                              float* __restrict__ x, int Bsz)
{
    int i = blockIdx.x * blockDim.x + threadIdx.x;
    int total = Bsz * NL * EDIM;
    if (i >= total) return;
    int d = i & 63;
    int rest = i >> 6;
    int f = rest % NL;
    int b = rest / NL;
    float v;
    if (f < 13) {
        int id = dfeat[b * 13 + f];
        v = dense_emb[((size_t)f * 50 + id) * EDIM + d];
    } else {
        int fs = f - 13;
        int id = sfeat[b * 26 + fs];
        v = sparse_emb[((size_t)fs * 50000 + id) * EDIM + d];
    }
    x[i] = v;
}

__global__ void ln_kernel(const float* __restrict__ x, const float* __restrict__ g,
                          const float* __restrict__ bta, float* __restrict__ y, int Mtot)
{
    int m = blockIdx.x * blockDim.x + threadIdx.x;
    if (m >= Mtot) return;
    const float* xr = x + (size_t)m * EDIM;
    float mean = 0.f;
    #pragma unroll
    for (int d = 0; d < EDIM; ++d) mean += xr[d];
    mean *= (1.f / EDIM);
    float var = 0.f;
    #pragma unroll
    for (int d = 0; d < EDIM; ++d) { float tt = xr[d] - mean; var += tt * tt; }
    var *= (1.f / EDIM);
    float inv = rsqrtf(var + 1e-5f);
    float* yr = y + (size_t)m * EDIM;
    #pragma unroll
    for (int d = 0; d < EDIM; ++d) yr[d] = (xr[d] - mean) * inv * g[d] + bta[d];
}

__global__ void sigA_kernel(const float* __restrict__ A, float* __restrict__ abuf)
{
    int i = blockIdx.x * blockDim.x + threadIdx.x;
    if (i >= EDIM * SDIM) return;
    abuf[i] = 1.f / (1.f + expf(-A[i]));
}

// One thread per (b, s).  h[d] = h[d]*a[d,s] + xp[b,l,d]*delta[b,l,s]; hsum = sum_d h[d].
__global__ void scan_kernel(const float* __restrict__ xp, const float* __restrict__ delta,
                            const float* __restrict__ abuf, float* __restrict__ hsum, int Bsz)
{
    int t = blockIdx.x * blockDim.x + threadIdx.x;
    if (t >= Bsz * SDIM) return;
    int b = t >> 4, s = t & 15;
    float a[EDIM], h[EDIM];
    #pragma unroll
    for (int d = 0; d < EDIM; ++d) { a[d] = abuf[d * SDIM + s]; h[d] = 0.f; }
    for (int l = 0; l < NL; ++l) {
        int m = b * NL + l;
        float dl = delta[(size_t)m * SDIM + s];
        const float* xr = xp + (size_t)m * EDIM;
        float hs = 0.f;
        #pragma unroll
        for (int d = 0; d < EDIM; ++d) {
            h[d] = h[d] * a[d] + xr[d] * dl;
            hs += h[d];
        }
        hsum[(size_t)m * SDIM + s] = hs;
    }
}

// y[m,d] = sum_s hsum[m,s]*Cw[d,s] + Dp[d]*xp[m,d]   (safe in-place when y == xp)
__global__ void combine_kernel(const float* __restrict__ hsum, const float* __restrict__ Cw,
                               const float* __restrict__ Dp, const float* __restrict__ xp,
                               float* __restrict__ y, int Mtot)
{
    int i = blockIdx.x * blockDim.x + threadIdx.x;
    if (i >= Mtot * EDIM) return;
    int m = i >> 6, d = i & 63;
    const float* hr = hsum + (size_t)m * SDIM;
    const float* cr = Cw + (size_t)d * SDIM;
    float acc = 0.f;
    #pragma unroll
    for (int s = 0; s < SDIM; ++s) acc += hr[s] * cr[s];
    y[i] = acc + Dp[d] * xp[i];
}

__global__ void head2_kernel(const float* __restrict__ h, const float* __restrict__ w2,
                             const float* __restrict__ b2, float* __restrict__ out, int Bsz)
{
    int b = blockIdx.x * blockDim.x + threadIdx.x;
    if (b >= Bsz) return;
    const float* hr = h + (size_t)b * HID;
    float acc = b2[0];
    #pragma unroll 8
    for (int j = 0; j < HID; ++j) acc += hr[j] * w2[j];
    out[b] = 1.f / (1.f + expf(-acc));
}

// ---------------- host orchestration ----------------

struct LayerP {
    const float *ln_g, *ln_b, *pin_w, *pin_b, *pout_w, *pout_b;
    const float *A, *Bw, *Cw, *Dp, *w1, *b1, *w2, *b2;
};

extern "C" void kernel_launch(void* const* d_in, const int* in_sizes, int n_in,
                              void* d_out, int out_size, void* d_ws, size_t ws_size,
                              hipStream_t stream)
{
    (void)ws_size; (void)n_in; (void)out_size;
    auto F = [&](int i) { return (const float*)d_in[i]; };

    // Determine flattening order of setup_inputs() dict.
    // mode 0: pure insertion order.
    // mode 1: jax-pytree alpha over whole dict (dense_feats, params-alpha, sparse_feats).
    // mode 2: outer insertion, params alphabetical.
    int mode = 0;
    if (in_sizes[0] == 8192 * 13) mode = 1;
    else if (in_sizes[1] == EDIM * SDIM) mode = 2;

    const float *dense_emb, *sparse_emb, *ow1, *ob1, *ow2, *ob2;
    const int *dfeat, *sfeat;
    LayerP lp[4];

    if (mode == 0) {
        dense_emb = F(0); sparse_emb = F(1);
        for (int l = 0; l < 4; ++l) {
            int b = 2 + 14 * l;
            lp[l] = { F(b+0), F(b+1), F(b+2), F(b+3), F(b+4), F(b+5),
                      F(b+6), F(b+7), F(b+8), F(b+9), F(b+10), F(b+11), F(b+12), F(b+13) };
        }
        ow1 = F(58); ob1 = F(59); ow2 = F(60); ob2 = F(61);
        dfeat = (const int*)d_in[62]; sfeat = (const int*)d_in[63];
    } else if (mode == 1) {
        dfeat = (const int*)d_in[0];
        dense_emb = F(1);
        for (int l = 0; l < 4; ++l) {
            int b = 2 + 14 * l;
            lp[l].A = F(b+0); lp[l].Bw = F(b+1); lp[l].Cw = F(b+2); lp[l].Dp = F(b+3);
            lp[l].b1 = F(b+4); lp[l].b2 = F(b+5); lp[l].w1 = F(b+6); lp[l].w2 = F(b+7);
            lp[l].ln_b = F(b+8); lp[l].ln_g = F(b+9);
            lp[l].pin_b = F(b+10); lp[l].pin_w = F(b+11);
            lp[l].pout_b = F(b+12); lp[l].pout_w = F(b+13);
        }
        ob1 = F(58); ob2 = F(59); ow1 = F(60); ow2 = F(61);
        sparse_emb = F(62); sfeat = (const int*)d_in[63];
    } else {
        dense_emb = F(0);
        for (int l = 0; l < 4; ++l) {
            int b = 1 + 14 * l;
            lp[l].A = F(b+0); lp[l].Bw = F(b+1); lp[l].Cw = F(b+2); lp[l].Dp = F(b+3);
            lp[l].b1 = F(b+4); lp[l].b2 = F(b+5); lp[l].w1 = F(b+6); lp[l].w2 = F(b+7);
            lp[l].ln_b = F(b+8); lp[l].ln_g = F(b+9);
            lp[l].pin_b = F(b+10); lp[l].pin_w = F(b+11);
            lp[l].pout_b = F(b+12); lp[l].pout_w = F(b+13);
        }
        ob1 = F(57); ob2 = F(58); ow1 = F(59); ow2 = F(60);
        sparse_emb = F(61);
        dfeat = (const int*)d_in[62]; sfeat = (const int*)d_in[63];
    }

    const int Bsz = 8192;
    const int M = Bsz * NL;             // 319488 token rows (multiple of 64)
    const int CHUNK = 65536;            // FFN M-chunk (multiple of 64; last chunk 57344 also is)

    // workspace layout (floats)
    float* ws = (float*)d_ws;
    size_t o = 0;
    float* X     = ws + o; o += (size_t)M * EDIM;   // activations
    float* XN    = ws + o; o += (size_t)M * EDIM;   // layernorm output (pout residual!)
    float* XP    = ws + o; o += (size_t)M * EDIM;   // pin output -> ymid (in place)
    float* DELTA = ws + o; o += (size_t)M * SDIM;
    float* HSUM  = ws + o; o += (size_t)M * SDIM;
    float* ABUF  = ws + o; o += 1024;
    float* H1    = ws + o; o += (size_t)CHUNK * HID; // FFN hidden chunk / head hidden

    dim3 blk(256);

    // embedding gather -> X [B, 39, 64]
    gather_kernel<<<dim3(((size_t)M * EDIM + 255) / 256), blk, 0, stream>>>(
        dense_emb, sparse_emb, dfeat, sfeat, X, Bsz);

    for (int l = 0; l < 4; ++l) {
        const LayerP& p = lp[l];
        // xn = LN(x)
        ln_kernel<<<dim3((M + 255) / 256), blk, 0, stream>>>(X, p.ln_g, p.ln_b, XN, M);
        // xp = xn @ pin_w^T + pin_b
        gemm_wmma<ACT_NONE><<<dim3(1, M / 64), blk, 0, stream>>>(
            XN, p.pin_w, p.pin_b, nullptr, XP, M, EDIM, EDIM);
        // delta = softplus(xp @ Bw^T)
        gemm_wmma<ACT_SOFTPLUS><<<dim3(1, M / 64), blk, 0, stream>>>(
            XP, p.Bw, nullptr, nullptr, DELTA, M, SDIM, EDIM);
        // a = sigmoid(A)
        sigA_kernel<<<dim3(4), blk, 0, stream>>>(p.A, ABUF);
        // sequential scan over L, parallel over (b, s)
        scan_kernel<<<dim3((Bsz * SDIM + 255) / 256), blk, 0, stream>>>(
            XP, DELTA, ABUF, HSUM, Bsz);
        // ymid = hsum @ Cw^T + Dp * xp   (in place into XP)
        combine_kernel<<<dim3(((size_t)M * EDIM + 255) / 256), blk, 0, stream>>>(
            HSUM, p.Cw, p.Dp, XP, XP, M);
        // x = ymid @ pout_w^T + pout_b + xn     (residual is the NORMED input!)
        gemm_wmma<ACT_NONE><<<dim3(1, M / 64), blk, 0, stream>>>(
            XP, p.pout_w, p.pout_b, XN, X, M, EDIM, EDIM);
        // FFN with residual, chunked over rows
        for (int m0 = 0; m0 < M; m0 += CHUNK) {
            int Mc = (M - m0 < CHUNK) ? (M - m0) : CHUNK;
            gemm_wmma<ACT_GELU><<<dim3(HID / 64, (Mc + 63) / 64), blk, 0, stream>>>(
                X + (size_t)m0 * EDIM, p.w1, p.b1, nullptr, H1, Mc, HID, EDIM);
            gemm_wmma<ACT_NONE><<<dim3(1, (Mc + 63) / 64), blk, 0, stream>>>(
                H1, p.w2, p.b2, X + (size_t)m0 * EDIM, X + (size_t)m0 * EDIM, Mc, EDIM, HID);
        }
    }

    // head: h = relu(flat[B, 2496] @ out_w1^T + out_b1)
    gemm_wmma<ACT_RELU><<<dim3(HID / 64, Bsz / 64), blk, 0, stream>>>(
        X, ow1, ob1, nullptr, H1, Bsz, HID, NL * EDIM);
    // out = sigmoid(h @ out_w2^T + out_b2)
    head2_kernel<<<dim3((Bsz + 255) / 256), blk, 0, stream>>>(
        H1, ow2, ob2, (float*)d_out, Bsz);
}
